// PointSlotAttention_62878321214017
// MI455X (gfx1250) — compile-verified, matrix operand order checked by reading, not run on hardware
//
#include <hip/hip_runtime.h>
#include <math.h>

#define BB 4
#define NN 32768
#define DD 128
#define SS 64
#define KK 16
#define TD 384
#define LNE 1e-5f

typedef __attribute__((ext_vector_type(16))) _Float16 v16h;
typedef __attribute__((ext_vector_type(8)))  float    v8f;

// ---------- wave32 helpers ----------
__device__ __forceinline__ float wave_sum(float v) {
#pragma unroll
  for (int off = 16; off >= 1; off >>= 1) v += __shfl_xor(v, off, 32);
  return v;
}

// gfx1250 16x16x32 f16 fragment: lanes 0-15 hold row m=lane with K {0..7,16..23}(+kbase),
// lanes 16-31 hold row m=lane-16 with K {8..15,24..31}(+kbase). Same striping used for
// B by indexing rows of W (since B = W^T, column n of B == row n of W, contiguous in k).
__device__ __forceinline__ v16h load_frag(const _Float16* base, int ld, int lane, int kbase) {
  v16h f;
  const int m  = lane & 15;
  const int hi = (lane >> 4) & 1;
  const _Float16* row = base + m * ld + kbase;
#pragma unroll
  for (int h = 0; h < 16; ++h) {
    const int k = (h < 8) ? (h + (hi ? 8 : 0)) : ((h - 8) + 16 + (hi ? 8 : 0));
    f[h] = row[k];
  }
  return f;
}

// ---------- 1) farthest point sampling (one block per batch, dist in VGPRs) ----------
__global__ void __launch_bounds__(1024) fps_kernel(const float* __restrict__ pos,
                                                   int* __restrict__ fps_idx,
                                                   float* __restrict__ slot_pos) {
  const int b = blockIdx.x, t = threadIdx.x;
  const float* pb = pos + (size_t)b * NN * 3;
  float dloc[32];
#pragma unroll
  for (int c = 0; c < 32; ++c) dloc[c] = 1e10f;
  __shared__ float s_val[32];
  __shared__ int   s_idx[32];
  __shared__ int   s_far;
  if (t == 0) s_far = 0;
  __syncthreads();
  for (int s = 0; s < SS; ++s) {
    const int far = s_far;
    if (t == 0) {
      fps_idx[b * SS + s] = far;
      slot_pos[(b * SS + s) * 3 + 0] = pb[far * 3 + 0];
      slot_pos[(b * SS + s) * 3 + 1] = pb[far * 3 + 1];
      slot_pos[(b * SS + s) * 3 + 2] = pb[far * 3 + 2];
    }
    const float cx = pb[far * 3 + 0], cy = pb[far * 3 + 1], cz = pb[far * 3 + 2];
    float bv = -1.0f; int bi = 0x7fffffff;
#pragma unroll
    for (int c = 0; c < 32; ++c) {
      const int i = t + c * 1024;
      const float dx = pb[i * 3 + 0] - cx, dy = pb[i * 3 + 1] - cy, dz = pb[i * 3 + 2] - cz;
      const float nd = fminf(dloc[c], dx * dx + dy * dy + dz * dz);
      dloc[c] = nd;
      if (nd > bv || (nd == bv && i < bi)) { bv = nd; bi = i; }
    }
#pragma unroll
    for (int off = 16; off >= 1; off >>= 1) {
      const float ov = __shfl_xor(bv, off, 32);
      const int   oi = __shfl_xor(bi, off, 32);
      if (ov > bv || (ov == bv && oi < bi)) { bv = ov; bi = oi; }
    }
    if ((t & 31) == 0) { s_val[t >> 5] = bv; s_idx[t >> 5] = bi; }
    __syncthreads();
    if (t == 0) {
      float mv = s_val[0]; int mi = s_idx[0];
      for (int w = 1; w < 32; ++w)
        if (s_val[w] > mv || (s_val[w] == mv && s_idx[w] < mi)) { mv = s_val[w]; mi = s_idx[w]; }
      s_far = mi;
    }
    __syncthreads();
  }
}

// ---------- 2) KNN: 16 argmin rounds per (b,s), distances recomputed from L2 ----------
__global__ void __launch_bounds__(1024) knn_kernel(const float* __restrict__ pos,
                                                   const float* __restrict__ slot_pos,
                                                   int* __restrict__ nn_idx) {
  const int g = blockIdx.x;              // b*SS + s
  const int b = g / SS, t = threadIdx.x;
  const float* pb = pos + (size_t)b * NN * 3;
  const float qx = slot_pos[g * 3 + 0], qy = slot_pos[g * 3 + 1], qz = slot_pos[g * 3 + 2];
  __shared__ float s_val[32];
  __shared__ int   s_idx[32];
  __shared__ int   sel[KK];
  for (int k = 0; k < KK; ++k) {
    float bv = 3.4e38f; int bi = 0x7fffffff;
    for (int i = t; i < NN; i += 1024) {
      bool skip = false;
      for (int j = 0; j < k; ++j) skip |= (sel[j] == i);
      if (skip) continue;
      const float dx = pb[i * 3 + 0] - qx, dy = pb[i * 3 + 1] - qy, dz = pb[i * 3 + 2] - qz;
      const float d = dx * dx + dy * dy + dz * dz;
      if (d < bv || (d == bv && i < bi)) { bv = d; bi = i; }
    }
#pragma unroll
    for (int off = 16; off >= 1; off >>= 1) {
      const float ov = __shfl_xor(bv, off, 32);
      const int   oi = __shfl_xor(bi, off, 32);
      if (ov < bv || (ov == bv && oi < bi)) { bv = ov; bi = oi; }
    }
    if ((t & 31) == 0) { s_val[t >> 5] = bv; s_idx[t >> 5] = bi; }
    __syncthreads();
    if (t == 0) {
      float mv = s_val[0]; int mi = s_idx[0];
      for (int w = 1; w < 32; ++w)
        if (s_val[w] < mv || (s_val[w] == mv && s_idx[w] < mi)) { mv = s_val[w]; mi = s_idx[w]; }
      sel[k] = mi; nn_idx[g * KK + k] = mi;
    }
    __syncthreads();
  }
}

// ---------- 3) column-sum constants (k_w/pe2_w collapse) ----------
__global__ void consts_kernel(const float* __restrict__ k_w, const float* __restrict__ k_b,
                              const float* __restrict__ pe2_w, const float* __restrict__ pe2_b,
                              float* __restrict__ kcol, float* __restrict__ pe2col,
                              float* __restrict__ scal) {
  const int c = threadIdx.x;
  float kc = 0.f, pc = 0.f;
  for (int d = 0; d < DD; ++d) { kc += k_w[d * DD + c]; pc += pe2_w[d * DD + c]; }
  kcol[c] = kc; pe2col[c] = pc;
  __shared__ float red[128];
  red[c] = k_b[c];
  for (int off = 64; off >= 1; off >>= 1) { __syncthreads(); if (c < off) red[c] += red[c + off]; }
  __syncthreads();
  if (c == 0) scal[0] = red[0];
  __syncthreads();
  red[c] = pe2_b[c];
  for (int off = 64; off >= 1; off >>= 1) { __syncthreads(); if (c < off) red[c] += red[c + off]; }
  __syncthreads();
  if (c == 0) scal[1] = red[0];
}

// ---------- 4) slots init = LN(inputs[fps rows]) ----------
__global__ void slots_init_kernel(const float* __restrict__ inputs, const int* __restrict__ fps_idx,
                                  const float* __restrict__ lw, const float* __restrict__ lb,
                                  float* __restrict__ slots) {
  const int g = blockIdx.x, b = g / SS, t = threadIdx.x;
  const int row = fps_idx[g];
  const float x = inputs[((size_t)b * NN + row) * DD + t];
  __shared__ float red[128];
  red[t] = x;
  for (int off = 64; off >= 1; off >>= 1) { __syncthreads(); if (t < off) red[t] += red[t + off]; }
  __syncthreads();
  const float mu = red[0] * (1.f / 128.f);
  __syncthreads();
  const float xc = x - mu;
  red[t] = xc * xc;
  for (int off = 64; off >= 1; off >>= 1) { __syncthreads(); if (t < off) red[t] += red[t + off]; }
  __syncthreads();
  const float rs = rsqrtf(red[0] * (1.f / 128.f) + LNE);
  slots[(size_t)g * DD + t] = xc * rs * lw[t] + lb[t];
}

// ---------- 5) gather + LN + V-projection (WMMA) + k row-sums ----------
__global__ void __launch_bounds__(256) gather_vproj_kernel(
    const float* __restrict__ inputs, const int* __restrict__ nn_idx,
    const float* __restrict__ v_w, const float* __restrict__ v_b,
    const float* __restrict__ lw, const float* __restrict__ lb,
    const float* __restrict__ kcol, const float* __restrict__ scal,
    float* __restrict__ v_nb, float* __restrict__ ksum) {
  __shared__ _Float16 sA[16][136];
  __shared__ _Float16 sW[128][136];
  const int t = threadIdx.x, lane = t & 31, wave = t >> 5;
  const int mbase = blockIdx.x * 16;
  for (int i = t; i < DD * DD; i += 256) sW[i >> 7][i & 127] = (_Float16)v_w[i];
  const float kb_sum = scal[0];
#pragma unroll
  for (int rr = 0; rr < 2; ++rr) {
    const int r = wave * 2 + rr;
    const int g = mbase + r;
    const int b = g / (SS * KK);
    const int row = nn_idx[g];
    const float4 xv = reinterpret_cast<const float4*>(inputs + ((size_t)b * NN + row) * DD)[lane];
    float s1 = xv.x + xv.y + xv.z + xv.w;
    float s2 = xv.x * xv.x + xv.y * xv.y + xv.z * xv.z + xv.w * xv.w;
    s1 = wave_sum(s1); s2 = wave_sum(s2);
    const float mu = s1 * (1.f / 128.f);
    const float rs = rsqrtf(s2 * (1.f / 128.f) - mu * mu + LNE);
    const float4 w4 = reinterpret_cast<const float4*>(lw)[lane];
    const float4 b4 = reinterpret_cast<const float4*>(lb)[lane];
    const float4 kc = reinterpret_cast<const float4*>(kcol)[lane];
    const float x0 = (xv.x - mu) * rs * w4.x + b4.x;
    const float x1 = (xv.y - mu) * rs * w4.y + b4.y;
    const float x2 = (xv.z - mu) * rs * w4.z + b4.z;
    const float x3 = (xv.w - mu) * rs * w4.w + b4.w;
    const int d0 = lane * 4;
    sA[r][d0 + 0] = (_Float16)x0; sA[r][d0 + 1] = (_Float16)x1;
    sA[r][d0 + 2] = (_Float16)x2; sA[r][d0 + 3] = (_Float16)x3;
    float ka = wave_sum(x0 * kc.x + x1 * kc.y + x2 * kc.z + x3 * kc.w);
    if (lane == 0) ksum[g] = ka + kb_sum;
  }
  __syncthreads();
  const int nb = wave * 16;
  v8f acc = {0.f, 0.f, 0.f, 0.f, 0.f, 0.f, 0.f, 0.f};
#pragma unroll
  for (int kb = 0; kb < DD; kb += 32) {
    v16h a  = load_frag(&sA[0][0], 136, lane, kb);
    v16h bm = load_frag(&sW[nb][0], 136, lane, kb);
    acc = __builtin_amdgcn_wmma_f32_16x16x32_f16(false, a, false, bm, (short)0, acc, false, false);
  }
  const int hi = (lane >> 4) & 1, nl = lane & 15, n = nb + nl;
  const float bias = v_b[n];
#pragma unroll
  for (int i = 0; i < 8; ++i)
    v_nb[(size_t)(mbase + i + (hi ? 8 : 0)) * DD + n] = acc[i] + bias;
}

// ---------- 6) positional-encoding row-sum -> logits (pe_sum - ksum) ----------
__global__ void pe_logit_kernel(const float* __restrict__ pos, const float* __restrict__ slot_pos,
                                const int* __restrict__ nn_idx,
                                const float* __restrict__ pe1_w, const float* __restrict__ pe1_b,
                                const float* __restrict__ pe2col, const float* __restrict__ scal,
                                const float* __restrict__ ksum, float* __restrict__ logit) {
  const int g = blockIdx.x, bs = g / KK, b = g / (SS * KK), t = threadIdx.x;
  const int row = nn_idx[g];
  const float rx = slot_pos[bs * 3 + 0] - pos[((size_t)b * NN + row) * 3 + 0];
  const float ry = slot_pos[bs * 3 + 1] - pos[((size_t)b * NN + row) * 3 + 1];
  const float rz = slot_pos[bs * 3 + 2] - pos[((size_t)b * NN + row) * 3 + 2];
  float h = fmaxf(pe1_w[t * 3 + 0] * rx + pe1_w[t * 3 + 1] * ry + pe1_w[t * 3 + 2] * rz + pe1_b[t], 0.f);
  __shared__ float red[128];
  red[t] = h * pe2col[t];
  for (int off = 64; off >= 1; off >>= 1) { __syncthreads(); if (t < off) red[t] += red[t + off]; }
  __syncthreads();
  if (t == 0) logit[g] = (red[0] + scal[1]) - ksum[g];
}

// ---------- 7) softmax over K + slot-normalize (q cancels in softmax) ----------
__global__ void __launch_bounds__(1024) attn_kernel(const float* __restrict__ logit,
                                                    float* __restrict__ att) {
  const int b = blockIdx.x, t = threadIdx.x, k = t & 15;
  const float lg = logit[b * SS * KK + t];
  float m = lg;
#pragma unroll
  for (int off = 8; off >= 1; off >>= 1) m = fmaxf(m, __shfl_xor(m, off, 32));
  const float e = __expf(lg - m);
  float sum = e;
#pragma unroll
  for (int off = 8; off >= 1; off >>= 1) sum += __shfl_xor(sum, off, 32);
  const float a = e / sum;
  __shared__ float sa[SS][KK];
  __shared__ float cs[KK];
  sa[t >> 4][k] = a;
  __syncthreads();
  if (t < KK) {
    float c = 0.f;
    for (int s2 = 0; s2 < SS; ++s2) c += sa[s2][t];
    cs[t] = c;
  }
  __syncthreads();
  att[b * SS * KK + t] = a / (cs[k] + 1e-6f);
}

// ---------- 8) upd = a . v_nb (iteration-invariant) ----------
__global__ void upd_kernel(const float* __restrict__ att, const float* __restrict__ v_nb,
                           float* __restrict__ upd) {
  const int bs = blockIdx.x, d = threadIdx.x;
  float u = 0.f;
#pragma unroll
  for (int k = 0; k < KK; ++k)
    u += att[bs * KK + k] * v_nb[(size_t)(bs * KK + k) * DD + d];
  upd[(size_t)bs * DD + d] = u;
}

// ---------- 9) generic 256x384x128 GEMM (WMMA) for GRU gates ----------
__global__ void __launch_bounds__(256) gemm384_kernel(const float* __restrict__ X,
                                                      const float* __restrict__ W,
                                                      const float* __restrict__ bias,
                                                      float* __restrict__ Y) {
  __shared__ _Float16 sA[16][136];
  __shared__ _Float16 sW[128][136];
  const int t = threadIdx.x, lane = t & 31, wave = t >> 5;
  const int mbase = blockIdx.x * 16, nchunk = blockIdx.y * 128;
  for (int i = t; i < 16 * DD; i += 256)
    sA[i >> 7][i & 127] = (_Float16)X[(size_t)(mbase + (i >> 7)) * DD + (i & 127)];
  for (int i = t; i < 128 * DD; i += 256)
    sW[i >> 7][i & 127] = (_Float16)W[(size_t)(nchunk + (i >> 7)) * DD + (i & 127)];
  __syncthreads();
  const int nb = wave * 16;
  v8f acc = {0.f, 0.f, 0.f, 0.f, 0.f, 0.f, 0.f, 0.f};
#pragma unroll
  for (int kb = 0; kb < DD; kb += 32) {
    v16h a  = load_frag(&sA[0][0], 136, lane, kb);
    v16h bm = load_frag(&sW[nb][0], 136, lane, kb);
    acc = __builtin_amdgcn_wmma_f32_16x16x32_f16(false, a, false, bm, (short)0, acc, false, false);
  }
  const int hi = (lane >> 4) & 1, nl = lane & 15, n = nchunk + nb + nl;
  const float bv = bias[n];
#pragma unroll
  for (int i = 0; i < 8; ++i)
    Y[(size_t)(mbase + i + (hi ? 8 : 0)) * TD + n] = acc[i] + bv;
}

// ---------- 10) GRU gate fuse (elementwise, in-place on slots) ----------
__global__ void gru_gate_kernel(const float* __restrict__ gi, const float* __restrict__ gh,
                                float* __restrict__ slots) {
  const int i = blockIdx.x * blockDim.x + threadIdx.x;
  const int bs = i >> 7, d = i & 127;
  const float r = 1.f / (1.f + __expf(-(gi[bs * TD + d] + gh[bs * TD + d])));
  const float z = 1.f / (1.f + __expf(-(gi[bs * TD + DD + d] + gh[bs * TD + DD + d])));
  const float n = tanhf(gi[bs * TD + 2 * DD + d] + r * gh[bs * TD + 2 * DD + d]);
  slots[i] = (1.f - z) * n + z * slots[i];
}

// ---------- 11) mlp1: LN + GEMM + ReLU (WMMA) ----------
__global__ void __launch_bounds__(256) mlp1_kernel(const float* __restrict__ slots,
                                                   const float* __restrict__ lw, const float* __restrict__ lb,
                                                   const float* __restrict__ W, const float* __restrict__ bias,
                                                   float* __restrict__ H) {
  __shared__ _Float16 sA[16][136];
  __shared__ _Float16 sW[128][136];
  const int t = threadIdx.x, lane = t & 31, wave = t >> 5;
  const int mbase = blockIdx.x * 16;
  for (int i = t; i < DD * DD; i += 256) sW[i >> 7][i & 127] = (_Float16)W[i];
#pragma unroll
  for (int rr = 0; rr < 2; ++rr) {
    const int r = wave * 2 + rr;
    const float4 xv = reinterpret_cast<const float4*>(slots + (size_t)(mbase + r) * DD)[lane];
    float s1 = xv.x + xv.y + xv.z + xv.w;
    float s2 = xv.x * xv.x + xv.y * xv.y + xv.z * xv.z + xv.w * xv.w;
    s1 = wave_sum(s1); s2 = wave_sum(s2);
    const float mu = s1 * (1.f / 128.f);
    const float rs = rsqrtf(s2 * (1.f / 128.f) - mu * mu + LNE);
    const float4 w4 = reinterpret_cast<const float4*>(lw)[lane];
    const float4 b4 = reinterpret_cast<const float4*>(lb)[lane];
    const int d0 = lane * 4;
    sA[r][d0 + 0] = (_Float16)((xv.x - mu) * rs * w4.x + b4.x);
    sA[r][d0 + 1] = (_Float16)((xv.y - mu) * rs * w4.y + b4.y);
    sA[r][d0 + 2] = (_Float16)((xv.z - mu) * rs * w4.z + b4.z);
    sA[r][d0 + 3] = (_Float16)((xv.w - mu) * rs * w4.w + b4.w);
  }
  __syncthreads();
  const int nb = wave * 16;
  v8f acc = {0.f, 0.f, 0.f, 0.f, 0.f, 0.f, 0.f, 0.f};
#pragma unroll
  for (int kb = 0; kb < DD; kb += 32) {
    v16h a  = load_frag(&sA[0][0], 136, lane, kb);
    v16h bm = load_frag(&sW[nb][0], 136, lane, kb);
    acc = __builtin_amdgcn_wmma_f32_16x16x32_f16(false, a, false, bm, (short)0, acc, false, false);
  }
  const int hi = (lane >> 4) & 1, nl = lane & 15, n = nb + nl;
  const float bv = bias[n];
#pragma unroll
  for (int i = 0; i < 8; ++i)
    H[(size_t)(mbase + i + (hi ? 8 : 0)) * DD + n] = fmaxf(acc[i] + bv, 0.f);
}

// ---------- 12) mlp2 + residual (WMMA) ----------
__global__ void __launch_bounds__(256) mlp2_kernel(const float* __restrict__ H,
                                                   const float* __restrict__ W,
                                                   const float* __restrict__ bias,
                                                   float* __restrict__ slots) {
  __shared__ _Float16 sA[16][136];
  __shared__ _Float16 sW[128][136];
  const int t = threadIdx.x, lane = t & 31, wave = t >> 5;
  const int mbase = blockIdx.x * 16;
  for (int i = t; i < 16 * DD; i += 256)
    sA[i >> 7][i & 127] = (_Float16)H[(size_t)(mbase + (i >> 7)) * DD + (i & 127)];
  for (int i = t; i < DD * DD; i += 256) sW[i >> 7][i & 127] = (_Float16)W[i];
  __syncthreads();
  const int nb = wave * 16;
  v8f acc = {0.f, 0.f, 0.f, 0.f, 0.f, 0.f, 0.f, 0.f};
#pragma unroll
  for (int kb = 0; kb < DD; kb += 32) {
    v16h a  = load_frag(&sA[0][0], 136, lane, kb);
    v16h bm = load_frag(&sW[nb][0], 136, lane, kb);
    acc = __builtin_amdgcn_wmma_f32_16x16x32_f16(false, a, false, bm, (short)0, acc, false, false);
  }
  const int hi = (lane >> 4) & 1, nl = lane & 15, n = nb + nl;
  const float bv = bias[n];
#pragma unroll
  for (int i = 0; i < 8; ++i) {
    const size_t o = (size_t)(mbase + i + (hi ? 8 : 0)) * DD + n;
    slots[o] = slots[o] + acc[i] + bv;
  }
}

extern "C" void kernel_launch(void* const* d_in, const int* in_sizes, int n_in,
                              void* d_out, int out_size, void* d_ws, size_t ws_size,
                              hipStream_t stream) {
  (void)in_sizes; (void)n_in; (void)out_size; (void)ws_size;
  // setup_inputs() order: inputs, pos, then params dict flattened.
  const float* inputs   = (const float*)d_in[0];
  const float* pos      = (const float*)d_in[1];
  // d_in[2]=q_w, d_in[3]=q_b : unused (q cancels inside K-softmax)
  const float* k_w      = (const float*)d_in[4];
  const float* k_b      = (const float*)d_in[5];
  const float* v_w      = (const float*)d_in[6];
  const float* v_b      = (const float*)d_in[7];
  const float* pe1_w    = (const float*)d_in[8];
  const float* pe1_b    = (const float*)d_in[9];
  const float* pe2_w    = (const float*)d_in[10];
  const float* pe2_b    = (const float*)d_in[11];
  const float* mlp1_w   = (const float*)d_in[12];
  const float* mlp1_b   = (const float*)d_in[13];
  const float* mlp2_w   = (const float*)d_in[14];
  const float* mlp2_b   = (const float*)d_in[15];
  const float* gru_w_ih = (const float*)d_in[16];
  const float* gru_w_hh = (const float*)d_in[17];
  const float* gru_b_ih = (const float*)d_in[18];
  const float* gru_b_hh = (const float*)d_in[19];
  const float* ln_in_w  = (const float*)d_in[20];
  const float* ln_in_b  = (const float*)d_in[21];
  // d_in[22..23] = ln_s_* : unused (only fed q projection)
  const float* ln_m_w   = (const float*)d_in[24];
  const float* ln_m_b   = (const float*)d_in[25];

  float* slots    = (float*)d_out;                 // (B,S,D)
  float* slot_pos = slots + BB * SS * DD;          // (B,S,3)

  char* base = (char*)d_ws; size_t o = 0;
  auto take = [&](size_t bytes) { void* p = base + o; o = (o + bytes + 255) & ~(size_t)255; return p; };
  int*   fps_idx = (int*)take(BB * SS * 4);
  int*   nn_idx  = (int*)take(BB * SS * KK * 4);
  float* kcol    = (float*)take(DD * 4);
  float* pe2col  = (float*)take(DD * 4);
  float* scal    = (float*)take(2 * 4);
  float* ksum    = (float*)take(BB * SS * KK * 4);
  float* logit   = (float*)take(BB * SS * KK * 4);
  float* att     = (float*)take(BB * SS * KK * 4);
  float* v_nb    = (float*)take((size_t)BB * SS * KK * DD * 4);   // 2 MB
  float* upd     = (float*)take((size_t)BB * SS * DD * 4);
  float* gi      = (float*)take((size_t)BB * SS * TD * 4);
  float* gh      = (float*)take((size_t)BB * SS * TD * 4);
  float* hbuf    = (float*)take((size_t)BB * SS * DD * 4);

  fps_kernel<<<BB, 1024, 0, stream>>>(pos, fps_idx, slot_pos);
  knn_kernel<<<BB * SS, 1024, 0, stream>>>(pos, slot_pos, nn_idx);
  consts_kernel<<<1, 128, 0, stream>>>(k_w, k_b, pe2_w, pe2_b, kcol, pe2col, scal);
  slots_init_kernel<<<BB * SS, 128, 0, stream>>>(inputs, fps_idx, ln_in_w, ln_in_b, slots);
  gather_vproj_kernel<<<(BB * SS * KK) / 16, 256, 0, stream>>>(
      inputs, nn_idx, v_w, v_b, ln_in_w, ln_in_b, kcol, scal, v_nb, ksum);
  pe_logit_kernel<<<BB * SS * KK, 128, 0, stream>>>(
      pos, slot_pos, nn_idx, pe1_w, pe1_b, pe2col, scal, ksum, logit);
  attn_kernel<<<BB, 1024, 0, stream>>>(logit, att);
  upd_kernel<<<BB * SS, 128, 0, stream>>>(att, v_nb, upd);
  gemm384_kernel<<<dim3(16, 3), 256, 0, stream>>>(upd, gru_w_ih, gru_b_ih, gi);  // loop-invariant

  for (int it = 0; it < 3; ++it) {
    gemm384_kernel<<<dim3(16, 3), 256, 0, stream>>>(slots, gru_w_hh, gru_b_hh, gh);
    gru_gate_kernel<<<(BB * SS * DD) / 256, 256, 0, stream>>>(gi, gh, slots);
    mlp1_kernel<<<16, 256, 0, stream>>>(slots, ln_m_w, ln_m_b, mlp1_w, mlp1_b, hbuf);
    mlp2_kernel<<<16, 256, 0, stream>>>(hbuf, mlp2_w, mlp2_b, slots);
  }
}
